// Model_15865609192079
// MI455X (gfx1250) — matrix operand order repaired and checked by reading, not compiled, not run on hardware
//
#include <hip/hip_runtime.h>

#define B_   256
#define L_   720
#define C_   321
#define W_   12
#define PLEN_ 96
#define S_   60
#define P_   8
#define R_   25

typedef __attribute__((ext_vector_type(16))) __bf16 v16bf;
typedef __attribute__((ext_vector_type(8)))  float  v8f;

union Frag {
  v16bf v;
  unsigned short h[16];
};

// round-half-up float->bf16 (1 VALU; the >>16 b16 store matches d16_hi)
static __device__ __forceinline__ unsigned short f2bf(float f) {
  unsigned int u = __float_as_uint(f) + 0x8000u;
  return (unsigned short)(u >> 16);
}

static __device__ __forceinline__ float fast_tanh(float x) {
#if __has_builtin(__builtin_amdgcn_tanhf)
  return __builtin_amdgcn_tanhf(x);
#elif __has_builtin(__builtin_amdgcn_tanh_f32)
  return __builtin_amdgcn_tanh_f32(x);
#else
  const float a = __builtin_fabsf(x);
  const float e = __builtin_amdgcn_exp2f(a * -2.8853900817779268f); // exp(-2a)
  const float r = __builtin_amdgcn_rcpf(1.0f + e);
  const float t = __builtin_fmaf(-2.0f * e, r, 1.0f);
  return __builtin_copysignf(t, x);
#endif
}

// ---------------------------------------------------------------------------
// Kernel 1: per-(b,c) mean over L and raw windowed segment projection.
// (mean correction is applied in kernel 2's preload, keeping this kernel a
// single streaming pass over x: reads coalesced over c, no global RMW.)
// seg stored [C][B][S] so kernel 2 reloads its (c, b-tile) slice contiguously.
// ---------------------------------------------------------------------------
__global__ void seg_mean_kernel(const float* __restrict__ x,
                                const float* __restrict__ segW,
                                float* __restrict__ seg_ws,
                                float* __restrict__ mean_ws) {
  const int b = blockIdx.x;
  const int c = threadIdx.x;
  if (c >= C_) return;

  float sw[W_];
#pragma unroll
  for (int w = 0; w < W_; ++w) sw[w] = segW[c * W_ + w];

  const float* xb = x + (size_t)b * L_ * C_ + c;
  float* segp = seg_ws + ((size_t)c * B_ + b) * S_;

  float msum = 0.f;
  for (int s = 0; s < S_; ++s) {
    float acc = 0.f;
#pragma unroll
    for (int w = 0; w < W_; ++w) {
      float v = xb[(size_t)(s * W_ + w) * C_];
      msum += v;
      acc = fmaf(v, sw[w], acc);
    }
    segp[s] = acc;                       // raw (uncorrected) segment
  }
  mean_ws[(size_t)b * C_ + c] = msum * (1.0f / (float)L_);
}

// ---------------------------------------------------------------------------
// Kernel 2: fused ESN recurrence, one wave32 per (channel, 16-batch tile).
//   H_new[16x25] = tanh(seg_s * Win + H @ Wres^T) via 2x wmma_f32_16x16x32_bf16
//   proj_W is planted as column n=25 of the B1 fragment, so acc1 (lanes sub==9)
//   yields z = H_s . proj_W inside the same WMMA.  acc[p] += out_W[p,s]*z.
// Epilogue: y[b,l,c] = sum_p acc[p]*f_W[l,p] + f_b[l] + mean[b,c]
// ---------------------------------------------------------------------------
__launch_bounds__(32)
__global__ void esn_kernel(const float* __restrict__ Win,
                           const float* __restrict__ Wres,
                           const float* __restrict__ outW,
                           const float* __restrict__ projW,
                           const float* __restrict__ fW,
                           const float* __restrict__ fb,
                           const float* __restrict__ segW,
                           const float* __restrict__ seg_ws,
                           const float* __restrict__ mean_ws,
                           float* __restrict__ out) {
  __shared__ __align__(8) unsigned short ldsH[16 * 32];  // H as bf16, stride 32
  __shared__ float ldsSegT[S_ * 16];                     // [s][m] transposed
  __shared__ float ldsZrow[16];
  __shared__ float ldsMean[16];
  __shared__ float ldsOutWT[S_ * P_];                    // [s][p] transposed
  __shared__ float ldsFW[PLEN_ * P_];
  __shared__ float ldsFB[PLEN_];

  const int lane  = threadIdx.x;       // 0..31, single wave per block
  const int c     = blockIdx.x % C_;
  const int btile = blockIdx.x / C_;
  const int b0    = btile * 16;
  const int half  = lane >> 4;         // 0 or 1
  const int sub   = lane & 15;

  // ---------------- preload ----------------
  for (int i = lane; i < 16 * 32; i += 32) ldsH[i] = 0;   // h0 = 0
  if (lane < 16) ldsMean[lane] = mean_ws[(size_t)(b0 + lane) * C_ + c];

  // sum of this channel's window weights (uniform -> scalar loads)
  float swsum = 0.f;
#pragma unroll
  for (int w = 0; w < W_; ++w) swsum += segW[c * W_ + w];

  __syncthreads();
  {
    // transpose + mean-correct the seg tile: seg_c = raw - mean[b]*swsum
    const float* src = seg_ws + ((size_t)c * B_ + b0) * S_;   // contiguous
    for (int i = lane; i < 16 * S_; i += 32) {
      const int m = i / S_, s = i % S_;
      ldsSegT[s * 16 + m] = fmaf(-ldsMean[m], swsum, src[i]);
    }
  }
  {
    const float* src = outW + (size_t)c * P_ * S_;
    for (int i = lane; i < P_ * S_; i += 32) {
      const int p = i / S_, s = i % S_;
      ldsOutWT[s * P_ + p] = src[i];
    }
  }
  for (int i = lane; i < PLEN_ * P_; i += 32) ldsFW[i] = fW[i];
  for (int i = lane; i < PLEN_; i += 32) ldsFB[i] = fb[i];

  // per-lane column constants (this lane owns output columns q0 and q1)
  const int q0 = sub, q1 = sub + 16;
  const float win0 = Win[(size_t)c * R_ + q0];
  const float win1 = (q1 < R_) ? Win[(size_t)c * R_ + q1] : 0.f;

  // B fragments: B[k][n] = Wres[c][n][k]; tile1 column n==25 carries proj_W.
  // ISA 16-bit B 32x16 layout: lane sub -> column n; VGPR j<4 -> K = 2j
  // (+8 upper half), j>=4 -> K = 16+2(j-4) (+8 upper); packs (K, K+1).
  Frag bf0, bf1;
  {
    const float* wres = Wres + (size_t)c * R_ * R_;
    const float* prj  = projW + (size_t)c * R_;
#pragma unroll
    for (int j = 0; j < 8; ++j) {
      const int k0 = ((j < 4) ? (2 * j) : (16 + 2 * (j - 4))) + (half ? 8 : 0);
      const int k1 = k0 + 1;
      const int n1 = q1;
      float a0 = (k0 < R_) ? wres[q0 * R_ + k0] : 0.f;
      float a1 = (k1 < R_) ? wres[q0 * R_ + k1] : 0.f;
      float b0v = 0.f, b1v = 0.f;
      if (n1 < R_) {
        b0v = (k0 < R_) ? wres[n1 * R_ + k0] : 0.f;
        b1v = (k1 < R_) ? wres[n1 * R_ + k1] : 0.f;
      } else if (n1 == R_) {                 // z-column: proj_W
        b0v = (k0 < R_) ? prj[k0] : 0.f;
        b1v = (k1 < R_) ? prj[k1] : 0.f;
      }
      bf0.h[2 * j]     = f2bf(a0);
      bf0.h[2 * j + 1] = f2bf(a1);
      bf1.h[2 * j]     = f2bf(b0v);
      bf1.h[2 * j + 1] = f2bf(b1v);
    }
  }

  float accout[P_];
#pragma unroll
  for (int p = 0; p < P_; ++p) accout[p] = 0.f;

  __syncthreads();

  const int M0 = half * 8;             // D rows owned by this half-wave
  const v8f cz = {0.f, 0.f, 0.f, 0.f, 0.f, 0.f, 0.f, 0.f};

  for (int s = 0; s < S_; ++s) {
    // A fragment from LDS H: 16-bit A 16x32 layout, M = lane&15;
    // VGPR j<4 -> K = 2j (+8 upper half), j>=4 -> K = 16+2(j-4) (+8 upper).
    Frag af;
#pragma unroll
    for (int j = 0; j < 8; ++j) {
      const int k = ((j < 4) ? (2 * j) : (16 + 2 * (j - 4))) + (half ? 8 : 0);
      af.h[2 * j]     = ldsH[sub * 32 + k];
      af.h[2 * j + 1] = ldsH[sub * 32 + k + 1];
    }

    v8f acc0 = __builtin_amdgcn_wmma_f32_16x16x32_bf16(
        false, af.v, false, bf0.v, (short)0, cz, false, false);
    v8f acc1 = __builtin_amdgcn_wmma_f32_16x16x32_bf16(
        false, af.v, false, bf1.v, (short)0, cz, false, false);

    // z of the PREVIOUS state appears in acc1 column 25 (lanes sub==9)
    if (sub == 9) {
#pragma unroll
      for (int i = 0; i < 8; ++i) ldsZrow[M0 + i] = acc1[i];
    }

#pragma unroll
    for (int i = 0; i < 8; ++i) {
      const int M = M0 + i;
      const float u = ldsSegT[s * 16 + M];          // merged broadcast loads
      const float h0 = fast_tanh(acc0[i] + u * win0);
      const float h1 = fast_tanh(acc1[i] + u * win1);  // padded cols -> tanh(0)=0
      ldsH[M * 32 + q0] = f2bf(h0);
      ldsH[M * 32 + q1] = f2bf(h1);
    }
    __syncthreads();

    // accumulate z_{s-1} (zero at s==0); all lanes run, upper half is a
    // harmless duplicate of rows 0..15 (same issue cost, no EXEC churn).
    {
      const int si = (s == 0) ? 0 : s - 1;
      const float z = ldsZrow[sub];
#pragma unroll
      for (int p = 0; p < P_; ++p)
        accout[p] = fmaf(z, ldsOutWT[si * P_ + p], accout[p]);
    }
    __syncthreads();
  }

  // tail: z of the final state (s = 59)
  {
    Frag af;
#pragma unroll
    for (int j = 0; j < 8; ++j) {
      const int k = ((j < 4) ? (2 * j) : (16 + 2 * (j - 4))) + (half ? 8 : 0);
      af.h[2 * j]     = ldsH[sub * 32 + k];
      af.h[2 * j + 1] = ldsH[sub * 32 + k + 1];
    }
    v8f accz = __builtin_amdgcn_wmma_f32_16x16x32_bf16(
        false, af.v, false, bf1.v, (short)0, cz, false, false);
    if (sub == 9) {
#pragma unroll
      for (int i = 0; i < 8; ++i) ldsZrow[M0 + i] = accz[i];
    }
    __syncthreads();
    {
      const float z = ldsZrow[sub];
#pragma unroll
      for (int p = 0; p < P_; ++p)
        accout[p] = fmaf(z, ldsOutWT[(S_ - 1) * P_ + p], accout[p]);
    }
  }

  // ---------------- epilogue: f_linear + bias + mean, write [B, 96, C] ----
  if (lane < 16) {
    const float meanv = ldsMean[lane];
    float* dst = out + (size_t)(b0 + lane) * PLEN_ * C_ + c;
    for (int l = 0; l < PLEN_; ++l) {
      float val = ldsFB[l] + meanv;
#pragma unroll
      for (int p = 0; p < P_; ++p)
        val = fmaf(accout[p], ldsFW[l * P_ + p], val);
      dst[(size_t)l * C_] = val;
    }
  }
}

// ---------------------------------------------------------------------------
extern "C" void kernel_launch(void* const* d_in, const int* in_sizes, int n_in,
                              void* d_out, int out_size, void* d_ws,
                              size_t ws_size, hipStream_t stream) {
  (void)in_sizes; (void)n_in; (void)out_size; (void)ws_size;
  const float* x     = (const float*)d_in[0];
  const float* segW  = (const float*)d_in[1];
  const float* Win   = (const float*)d_in[2];
  const float* Wres  = (const float*)d_in[3];
  const float* outW  = (const float*)d_in[4];
  const float* projW = (const float*)d_in[5];
  const float* fW    = (const float*)d_in[6];
  const float* fb    = (const float*)d_in[7];
  float* out = (float*)d_out;

  float* seg_ws  = (float*)d_ws;                       // [C][B][S] = 19.7 MB
  float* mean_ws = seg_ws + (size_t)C_ * B_ * S_;      // [B][C]

  seg_mean_kernel<<<B_, 352, 0, stream>>>(x, segW, seg_ws, mean_ws);
  esn_kernel<<<C_ * (B_ / 16), 32, 0, stream>>>(Win, Wres, outW, projW, fW, fb,
                                                segW, seg_ws, mean_ws, out);
}
